// DeepseekV2Attention_20366734918038
// MI455X (gfx1250) — compile-verified, matrix operand order checked by reading, not run on hardware
//
#include <hip/hip_runtime.h>
#include <cstddef>
#include <cstdint>

typedef __bf16 bf16;
typedef __attribute__((ext_vector_type(16))) __bf16 v16bf;
typedef __attribute__((ext_vector_type(8)))  __bf16 v8bf;
typedef __attribute__((ext_vector_type(8)))  float  v8f;
typedef __attribute__((ext_vector_type(4)))  int    v4i;

union F8 { v8f v; float f[8]; };

namespace {
constexpr int   kS     = 2048;
constexpr int   kH     = 4096;
constexpr int   kNH    = 32;
constexpr int   kNOPE  = 128;
constexpr int   kROPE  = 64;
constexpr int   kVD    = 128;
constexpr int   kQHD   = 192;     // NOPE + ROPE
constexpr int   kQLR   = 1536;
constexpr int   kKVLR  = 512;
constexpr int   kCKV   = 576;     // KVLR + ROPE
constexpr int   kQOUT  = 6144;    // NH * QHD
constexpr int   kKVOUT = 8192;    // NH * (NOPE + VD)
constexpr float kScale = 0.07216878364870323f;  // 192^-0.5
constexpr float kRopeC = 0.28782313662425572f;  // ln(10000)/32
}

// ---- CDNA5 async global->LDS copy (16B per lane), ASYNCcnt-tracked ----------
#if defined(__HIP_DEVICE_COMPILE__) && __has_builtin(__builtin_amdgcn_global_load_async_to_lds_b128)
#define USE_ASYNC_LDS 1
#else
#define USE_ASYNC_LDS 0
#endif

#if USE_ASYNC_LDS
typedef __attribute__((address_space(1))) v4i as1_v4i;
typedef __attribute__((address_space(3))) v4i as3_v4i;
#endif

__device__ __forceinline__ void async_copy16(const bf16* g, bf16* l) {
#if USE_ASYNC_LDS
  // generic->AS1: identity on global addresses; generic->AS3: low 32 bits are
  // the wave-relative LDS offset.
  __builtin_amdgcn_global_load_async_to_lds_b128(
      (as1_v4i*)(uintptr_t)g,
      (as3_v4i*)(uint32_t)(uintptr_t)l, 0, 0);
#else
  *(v8bf*)l = *(const v8bf*)g;
#endif
}

__device__ __forceinline__ void wait_async() {
#if USE_ASYNC_LDS
#if __has_builtin(__builtin_amdgcn_s_wait_asynccnt)
  __builtin_amdgcn_s_wait_asynccnt(0);
#else
  asm volatile("s_wait_asynccnt 0" ::: "memory");
#endif
#endif
}

// Load one 16-bit WMMA A/B fragment half-pair: elements 0..7 from p[0..7],
// elements 8..15 from p[16..23]  (caller supplies p = &row[kbase + (lane>>4)*8]).
__device__ __forceinline__ v16bf load_frag(const bf16* p) {
  v16bf r;
  *(v8bf*)&r         = *(const v8bf*)p;
  *(((v8bf*)&r) + 1) = *(const v8bf*)(p + 16);
  return r;
}

__global__ void cvt_f32_to_bf16(const float* __restrict__ x, bf16* __restrict__ y, int n) {
  int i = blockIdx.x * blockDim.x + threadIdx.x;
  if (i < n) y[i] = (bf16)x[i];
}

// One block per row: var = mean(x^2); out = bf16(w * x * rsqrt(var + eps))
__global__ __launch_bounds__(256) void rmsnorm_bf16(const float* __restrict__ x,
                                                    const float* __restrict__ w,
                                                    bf16* __restrict__ out,
                                                    int D, int stride) {
  __shared__ float red[256];
  const int row = blockIdx.x;
  const int tid = threadIdx.x;
  const float* xr = x + row * stride;
  float s = 0.f;
  for (int d = tid; d < D; d += 256) { float v = xr[d]; s += v * v; }
  red[tid] = s;
  __syncthreads();
  for (int o = 128; o > 0; o >>= 1) {
    if (tid < o) red[tid] += red[tid + o];
    __syncthreads();
  }
  const float r = rsqrtf(red[0] / (float)D + 1e-6f);
  bf16* orow = out + row * D;
  for (int d = tid; d < D; d += 256) orow[d] = (bf16)(w[d] * (xr[d] * r));
}

// C[M,N] (f32) = A[M,K] (bf16 row-major) x B[N,K]^T (bf16 row-major)
// 256 threads = 8 waves in a 4x2 grid. Wave tile = 32 x (16*NT).
// Block tile = 128 x (32*NT).  NT=4 -> 128x128, NT=2 -> 128x64.
// Async-LDS double buffering: copy tile k+1 while computing tile k.
template <int NT>
__global__ __launch_bounds__(256) void gemm_bf16_nt(const bf16* __restrict__ A,
                                                    const bf16* __restrict__ B,
                                                    float* __restrict__ C,
                                                    int N, int K) {
  constexpr int BM = 128, BK = 32, PAD = 8;
  constexpr int BN = 32 * NT;
  __shared__ __align__(16) bf16 As[2][BM][BK + PAD];  // 80B row stride: conflict-free
  __shared__ __align__(16) bf16 Bs[2][BN][BK + PAD];
  const int tid   = threadIdx.x;
  const int lane  = tid & 31;
  const int wv    = tid >> 5;
  const int wm    = wv >> 1, wn = wv & 1;
  const int m0    = blockIdx.y * BM, n0 = blockIdx.x * BN;
  const int half8 = (lane >> 4) * 8;
  const int colL  = lane & 15;

  F8 acc[2][NT];
#pragma unroll
  for (int i = 0; i < 2; ++i)
#pragma unroll
    for (int j = 0; j < NT; ++j)
#pragma unroll
      for (int r = 0; r < 8; ++r) acc[i][j].f[r] = 0.f;

  auto issue_tile = [&](int k0, int buf) {
#pragma unroll
    for (int c = tid; c < BM * 4; c += 256) {      // A tile: BM x 32 in 16B chunks
      const int r = c >> 2, cc = (c & 3) * 8;
      async_copy16(&A[(m0 + r) * K + k0 + cc], &As[buf][r][cc]);
    }
#pragma unroll
    for (int c = tid; c < BN * 4; c += 256) {      // B tile: BN x 32
      const int r = c >> 2, cc = (c & 3) * 8;
      async_copy16(&B[(n0 + r) * K + k0 + cc], &Bs[buf][r][cc]);
    }
  };

  const int nk = K / BK;
  issue_tile(0, 0);
  for (int kt = 0; kt < nk; ++kt) {
    const int buf = kt & 1;
    wait_async();        // this wave's copies for tile kt are in LDS
    __syncthreads();     // ...and everyone else's; prev buffer fully consumed
    if (kt + 1 < nk) issue_tile((kt + 1) * BK, buf ^ 1);  // overlap with compute

    v16bf a[2], b[NT];
#pragma unroll
    for (int mi = 0; mi < 2; ++mi)
      a[mi] = load_frag(&As[buf][wm * 32 + mi * 16 + colL][half8]);
#pragma unroll
    for (int ni = 0; ni < NT; ++ni)
      b[ni] = load_frag(&Bs[buf][wn * 16 * NT + ni * 16 + colL][half8]);
#pragma unroll
    for (int mi = 0; mi < 2; ++mi)
#pragma unroll
      for (int ni = 0; ni < NT; ++ni)
        acc[mi][ni].v = __builtin_amdgcn_wmma_f32_16x16x32_bf16(
            false, a[mi], false, b[ni], (short)0, acc[mi][ni].v, false, false);
  }

#pragma unroll
  for (int mi = 0; mi < 2; ++mi)
#pragma unroll
    for (int ni = 0; ni < NT; ++ni) {
      const int col = n0 + wn * 16 * NT + ni * 16 + colL;
#pragma unroll
      for (int r = 0; r < 8; ++r) {
        const int row = m0 + wm * 32 + mi * 16 + r + half8;
        C[row * N + col] = acc[mi][ni].f[r];
      }
    }
}

// q_f32[S, NH*192] -> Q_bf16[NH][S][192], RoPE on last 64 dims, softmax scale folded in.
__global__ __launch_bounds__(256) void build_q(const float* __restrict__ q, bf16* __restrict__ Q) {
  const int s = blockIdx.x;
  const float* qr = q + s * kQOUT;
  for (int e = threadIdx.x; e < kNH * kQHD; e += 256) {
    const int h = e / kQHD, d = e - h * kQHD;
    float val;
    if (d < kNOPE) {
      val = qr[h * kQHD + d];
    } else {
      const int i = d - kNOPE;
      const int j = i & 31;
      const float inv = __expf(-(float)j * kRopeC);
      const float ang = (float)s * inv;
      const float xi = qr[h * kQHD + kNOPE + i];
      const float xo = (i < 32) ? -qr[h * kQHD + kNOPE + i + 32]
                                :  qr[h * kQHD + kNOPE + i - 32];
      val = xi * __cosf(ang) + xo * __sinf(ang);
    }
    Q[(h * kS + s) * kQHD + d] = (bf16)(val * kScale);
  }
}

// kv_f32[S, NH*256] + ckv_f32[S,576] -> K_bf16[NH][S][192] (nope | roped k_pe broadcast),
// V_bf16 transposed per head: [NH][128][S]
__global__ __launch_bounds__(256) void build_kv(const float* __restrict__ kv,
                                                const float* __restrict__ ckv,
                                                bf16* __restrict__ K,
                                                bf16* __restrict__ Vt) {
  const int s = blockIdx.x;
  __shared__ float kpe[kROPE];
  if (threadIdx.x < kROPE) {
    const int i = threadIdx.x;
    const int j = i & 31;
    const float inv = __expf(-(float)j * kRopeC);
    const float ang = (float)s * inv;
    const float xi = ckv[s * kCKV + kKVLR + i];
    const float xo = (i < 32) ? -ckv[s * kCKV + kKVLR + i + 32]
                              :  ckv[s * kCKV + kKVLR + i - 32];
    kpe[i] = xi * __cosf(ang) + xo * __sinf(ang);
  }
  __syncthreads();
  const float* kvr = kv + s * kKVOUT;
  for (int e = threadIdx.x; e < kNH * kQHD; e += 256) {
    const int h = e / kQHD, d = e - h * kQHD;
    const float val = (d < kNOPE) ? kvr[h * 256 + d] : kpe[d - kNOPE];
    K[(h * kS + s) * kQHD + d] = (bf16)val;
  }
  for (int e = threadIdx.x; e < kNH * kVD; e += 256) {
    const int h = e / kVD, d = e - h * kVD;
    Vt[(h * kVD + d) * kS + s] = (bf16)kvr[h * 256 + kNOPE + d];
  }
}

// Flash attention: one block = (head, 64 q rows); 4 waves, each owns 16 q rows.
// Q fragments live in registers; K / V^T async-staged in LDS; online softmax.
__global__ __launch_bounds__(128) void mla_attention(const bf16* __restrict__ Q,
                                                     const bf16* __restrict__ K,
                                                     const bf16* __restrict__ Vt,
                                                     bf16* __restrict__ O) {
  __shared__ __align__(16) bf16 Ks[64][200];   // 192 + 8 pad  (25600 B)
  __shared__ __align__(16) bf16 Vts[128][72];  // 64 keys + 8 pad (18432 B)
  __shared__ __align__(16) bf16 Ps[64][72];    // 64 + 8 pad   (9216 B)

  const int h     = blockIdx.y;
  const int q0    = blockIdx.x * 64;
  const int tid   = threadIdx.x;
  const int lane  = tid & 31;
  const int wv    = tid >> 5;          // wave -> q rows [wv*16, wv*16+16)
  const int colL  = lane & 15;
  const int half8 = (lane >> 4) * 8;

  const bf16* Qh = Q  + (size_t)h * kS  * kQHD;
  const bf16* Kh = K  + (size_t)h * kS  * kQHD;
  const bf16* Vh = Vt + (size_t)h * kVD * kS;

  // Q fragments in registers: 16x192 per wave = 6 k-steps
  v16bf qf[6];
#pragma unroll
  for (int kk = 0; kk < 6; ++kk)
    qf[kk] = load_frag(&Qh[(q0 + wv * 16 + colL) * kQHD + kk * 32 + half8]);

  float m_i[8], l_i[8];
  F8 oAcc[8];
#pragma unroll
  for (int r = 0; r < 8; ++r) { m_i[r] = -1e30f; l_i[r] = 0.f; }
#pragma unroll
  for (int nt = 0; nt < 8; ++nt)
#pragma unroll
    for (int r = 0; r < 8; ++r) oAcc[nt].f[r] = 0.f;

  const int nkt = blockIdx.x + 1;  // causal: kv tiles 0..q0/64
  for (int kt = 0; kt < nkt; ++kt) {
    const int k0 = kt * 64;
#pragma unroll
    for (int i = 0; i < 12; ++i) {   // K tile 64x192: 1536 16B-chunks / 128 thr
      const int c = tid + i * 128;
      const int r = c / 24, cc = (c % 24) * 8;
      async_copy16(&Kh[(k0 + r) * kQHD + cc], &Ks[r][cc]);
    }
#pragma unroll
    for (int i = 0; i < 8; ++i) {    // V^T tile 128x64: 1024 chunks / 128 thr
      const int c = tid + i * 128;
      const int r = c >> 3, cc = (c & 7) * 8;
      async_copy16(&Vh[r * kS + k0 + cc], &Vts[r][cc]);
    }
    wait_async();
    __syncthreads();

    // S = Q K^T : 16x64, over d=192
    F8 sc[4];
#pragma unroll
    for (int n = 0; n < 4; ++n)
#pragma unroll
      for (int r = 0; r < 8; ++r) sc[n].f[r] = 0.f;
#pragma unroll
    for (int kk = 0; kk < 6; ++kk) {
      const v16bf a = qf[kk];
#pragma unroll
      for (int n = 0; n < 4; ++n) {
        const v16bf b = load_frag(&Ks[n * 16 + colL][kk * 32 + half8]);
        sc[n].v = __builtin_amdgcn_wmma_f32_16x16x32_bf16(
            false, a, false, b, (short)0, sc[n].v, false, false);
      }
    }

    if (kt == nkt - 1) {  // diagonal tile -> causal mask
#pragma unroll
      for (int n = 0; n < 4; ++n) {
        const int key = k0 + n * 16 + colL;
#pragma unroll
        for (int r = 0; r < 8; ++r) {
          const int qr = q0 + wv * 16 + r + half8;
          if (key > qr) sc[n].f[r] = -1e30f;
        }
      }
    }

    // online softmax (row r lives in 16 lanes of one half-wave)
#pragma unroll
    for (int r = 0; r < 8; ++r) {
      float v = sc[0].f[r];
#pragma unroll
      for (int n = 1; n < 4; ++n) v = fmaxf(v, sc[n].f[r]);
#pragma unroll
      for (int msk = 1; msk < 16; msk <<= 1) v = fmaxf(v, __shfl_xor(v, msk, 32));
      const float mNew  = fmaxf(m_i[r], v);
      const float alpha = __expf(m_i[r] - mNew);
      m_i[r] = mNew;
      float s = 0.f;
#pragma unroll
      for (int n = 0; n < 4; ++n) {
        const float p = __expf(sc[n].f[r] - mNew);
        sc[n].f[r] = p;
        s += p;
      }
#pragma unroll
      for (int msk = 1; msk < 16; msk <<= 1) s += __shfl_xor(s, msk, 32);
      l_i[r] = l_i[r] * alpha + s;
#pragma unroll
      for (int nt = 0; nt < 8; ++nt) oAcc[nt].f[r] *= alpha;
    }

    // P -> LDS (re-layout C-frag -> A-frag)
#pragma unroll
    for (int n = 0; n < 4; ++n)
#pragma unroll
      for (int r = 0; r < 8; ++r)
        Ps[wv * 16 + r + half8][n * 16 + colL] = (bf16)sc[n].f[r];
    __syncthreads();

    // O += P @ V : 16x128 over 64 keys
#pragma unroll
    for (int kk = 0; kk < 2; ++kk) {
      const v16bf a = load_frag(&Ps[wv * 16 + colL][kk * 32 + half8]);
#pragma unroll
      for (int nt = 0; nt < 8; ++nt) {
        const v16bf b = load_frag(&Vts[nt * 16 + colL][kk * 32 + half8]);
        oAcc[nt].v = __builtin_amdgcn_wmma_f32_16x16x32_bf16(
            false, a, false, b, (short)0, oAcc[nt].v, false, false);
      }
    }
    __syncthreads();
  }

  // normalize + store to [S, NH*VD] (bf16, feeds final GEMM)
#pragma unroll
  for (int r = 0; r < 8; ++r) {
    const float inv = 1.f / l_i[r];
    const int row = q0 + wv * 16 + r + half8;
#pragma unroll
    for (int nt = 0; nt < 8; ++nt)
      O[row * (kNH * kVD) + h * kVD + nt * 16 + colL] =
          (bf16)(oAcc[nt].f[r] * inv);
  }
}

extern "C" void kernel_launch(void* const* d_in, const int* in_sizes, int n_in,
                              void* d_out, int out_size, void* d_ws, size_t ws_size,
                              hipStream_t stream) {
  (void)in_sizes; (void)n_in; (void)out_size; (void)ws_size;
  const float* hidden = (const float*)d_in[0];
  // d_in[1] = attention_mask: causal, computed analytically, unused
  const float* Wqa   = (const float*)d_in[2];
  const float* qa_w  = (const float*)d_in[3];
  const float* Wqb   = (const float*)d_in[4];
  const float* Wkva  = (const float*)d_in[5];
  const float* kva_w = (const float*)d_in[6];
  const float* Wkvb  = (const float*)d_in[7];
  const float* Wo    = (const float*)d_in[8];
  float* out = (float*)d_out;

  char* ws = (char*)d_ws;
  size_t off = 0;
  auto alloc = [&](size_t bytes) -> char* {
    char* p = ws + off;
    off += (bytes + 255) & ~(size_t)255;
    return p;
  };

  bf16*  hid_b  = (bf16*) alloc((size_t)kS * kH * 2);
  bf16*  Wqa_b  = (bf16*) alloc((size_t)kQLR * kH * 2);
  bf16*  Wqb_b  = (bf16*) alloc((size_t)kQOUT * kQLR * 2);
  bf16*  Wkva_b = (bf16*) alloc((size_t)kCKV * kH * 2);
  bf16*  Wkvb_b = (bf16*) alloc((size_t)kKVOUT * kKVLR * 2);
  bf16*  Wo_b   = (bf16*) alloc((size_t)kH * kH * 2);
  float* qlat_f = (float*)alloc((size_t)kS * kQLR * 4);
  float* ckv_f  = (float*)alloc((size_t)kS * kCKV * 4);
  bf16*  qlat_b = (bf16*) alloc((size_t)kS * kQLR * 2);
  bf16*  ckv_b  = (bf16*) alloc((size_t)kS * kKVLR * 2);
  float* q_f    = (float*)alloc((size_t)kS * kQOUT * 4);
  float* kv_f   = (float*)alloc((size_t)kS * kKVOUT * 4);
  bf16*  Qb     = (bf16*) alloc((size_t)kNH * kS * kQHD * 2);
  bf16*  Kb     = (bf16*) alloc((size_t)kNH * kS * kQHD * 2);
  bf16*  Vtb    = (bf16*) alloc((size_t)kNH * kVD * kS * 2);
  bf16*  attn_b = (bf16*) alloc((size_t)kS * kH * 2);

  auto cvt = [&](const float* x, bf16* y, size_t n) {
    cvt_f32_to_bf16<<<(unsigned)((n + 255) / 256), 256, 0, stream>>>(x, y, (int)n);
  };
  cvt(hidden, hid_b,  (size_t)kS * kH);
  cvt(Wqa,    Wqa_b,  (size_t)kQLR * kH);
  cvt(Wqb,    Wqb_b,  (size_t)kQOUT * kQLR);
  cvt(Wkva,   Wkva_b, (size_t)kCKV * kH);
  cvt(Wkvb,   Wkvb_b, (size_t)kKVOUT * kKVLR);
  cvt(Wo,     Wo_b,   (size_t)kH * kH);

  // 128x128 tiles (NT=4) for the big GEMMs; 128x64 (NT=2) for N=576.
  gemm_bf16_nt<4><<<dim3(kQLR / 128,  kS / 128), 256, 0, stream>>>(hid_b,  Wqa_b,  qlat_f, kQLR,  kH);
  gemm_bf16_nt<2><<<dim3(kCKV / 64,   kS / 128), 256, 0, stream>>>(hid_b,  Wkva_b, ckv_f,  kCKV,  kH);
  rmsnorm_bf16<<<kS, 256, 0, stream>>>(qlat_f, qa_w,  qlat_b, kQLR,  kQLR);
  rmsnorm_bf16<<<kS, 256, 0, stream>>>(ckv_f,  kva_w, ckv_b,  kKVLR, kCKV);
  gemm_bf16_nt<4><<<dim3(kQOUT / 128, kS / 128), 256, 0, stream>>>(qlat_b, Wqb_b,  q_f,  kQOUT,  kQLR);
  gemm_bf16_nt<4><<<dim3(kKVOUT / 128, kS / 128), 256, 0, stream>>>(ckv_b, Wkvb_b, kv_f, kKVOUT, kKVLR);
  build_q <<<kS, 256, 0, stream>>>(q_f, Qb);
  build_kv<<<kS, 256, 0, stream>>>(kv_f, ckv_f, Kb, Vtb);
  mla_attention<<<dim3(kS / 64, kNH), 128, 0, stream>>>(Qb, Kb, Vtb, attn_b);
  gemm_bf16_nt<4><<<dim3(kH / 128, kS / 128), 256, 0, stream>>>(attn_b, Wo_b, out, kH, kH);
}
